// MultiHeadedSelfAttention_78726750536054
// MI455X (gfx1250) — matrix operand order repaired and checked by reading, not compiled, not run on hardware
//
#include <hip/hip_runtime.h>
#include <hip/hip_bf16.h>

typedef __bf16 bf16_t;
typedef __attribute__((ext_vector_type(16))) __bf16 v16bf;
typedef __attribute__((ext_vector_type(8)))  __bf16 v8bf;
typedef __attribute__((ext_vector_type(2)))  __bf16 v2bf;
typedef __attribute__((ext_vector_type(8)))  float  v8f;

#define BATCH  8
#define CDIM   512
#define NPIX   1024
#define NHEADS 4
#define DQK    128   // total q/k channels
#define DQH    32    // per-head q/k dim
#define DVH    128   // per-head v dim
#define INV_SCALE 0.0883883476483184f   // 1/sqrt(C/heads) = 1/sqrt(128)

// compiler-level scheduling fence: memory ops cannot be hoisted across
#define SCHED_FENCE() asm volatile("" ::: "memory")

// ---- WMMA operand loaders (bf16, 16x16x32) ---------------------------------
// A-matrix 16x32 (MxK): lane row M = lane&15; element e -> K = e + 16*hi.
//                       Two 16B loads.
__device__ __forceinline__ v16bf load_a(const bf16_t* __restrict__ row, int hi) {
    const v8bf lo = *reinterpret_cast<const v8bf*>(row + 16 * hi);
    const v8bf hv = *reinterpret_cast<const v8bf*>(row + 16 * hi + 8);
    v16bf r;
#pragma unroll
    for (int i = 0; i < 8; ++i) { r[i] = lo[i]; r[i + 8] = hv[i]; }
    return r;
}
// B-matrix 32x16 (KxN): lane col N = lane&15; element e -> K = e + 16*hi.
__device__ __forceinline__ v16bf load_b(const bf16_t* __restrict__ col, int hi) {
    const v8bf lo = *reinterpret_cast<const v8bf*>(col + 16 * hi);
    const v8bf hv = *reinterpret_cast<const v8bf*>(col + 16 * hi + 8);
    v16bf r;
#pragma unroll
    for (int i = 0; i < 8; ++i) { r[i] = lo[i]; r[i + 8] = hv[i]; }
    return r;
}

// ---- conversion kernels ----------------------------------------------------
// x [B,C,N] f32 -> xT [B,N,C] bf16 ; two c-elements per thread, packed b32 store
__global__ void cvt_x_transpose(const float* __restrict__ x, bf16_t* __restrict__ xT) {
    size_t i = (size_t)blockIdx.x * blockDim.x + threadIdx.x;
    if (i >= (size_t)BATCH * (CDIM / 2) * NPIX) return;
    size_t b = i / ((size_t)(CDIM / 2) * NPIX);
    size_t rem = i % ((size_t)(CDIM / 2) * NPIX);
    size_t c2 = rem / NPIX, n = rem % NPIX;
    float f0 = x[(b * CDIM + 2 * c2) * NPIX + n];
    float f1 = x[(b * CDIM + 2 * c2 + 1) * NPIX + n];
    v2bf p = { (bf16_t)f0, (bf16_t)f1 };
    *reinterpret_cast<v2bf*>(xT + (b * NPIX + n) * CDIM + 2 * c2) = p;
}

// f32 -> bf16 with optional scale (used to fold 1/sqrt(dk) into Wq)
__global__ void cvt_flat(const float* __restrict__ in, bf16_t* __restrict__ out,
                         int n, float scale) {
    int i = blockIdx.x * blockDim.x + threadIdx.x;
    if (i < n) out[i] = (bf16_t)(in[i] * scale);
}

// ---- projection GEMM: out[o,n] = sum_c W[o,c] * xT[n,c] + bias[o]*bscale ---
template <bool TRANSPOSED>
__global__ void __launch_bounds__(32, 1)
proj_wmma(const bf16_t* __restrict__ Wb, const float* __restrict__ bias,
          const bf16_t* __restrict__ xT, bf16_t* __restrict__ out,
          int O, float bscale) {
    const int lane = threadIdx.x & 31;
    const int lm = lane & 15, hi = lane >> 4;
    const int n0 = blockIdx.x * 16, o0 = blockIdx.y * 16, b = blockIdx.z;
    const bf16_t* wrow = Wb + (size_t)(o0 + lm) * CDIM;
    const bf16_t* xrow = xT + ((size_t)b * NPIX + n0 + lm) * CDIM;
    v8f acc = {};
#pragma unroll 4
    for (int c0 = 0; c0 < CDIM; c0 += 32) {
        v16bf a  = load_a(wrow + c0, hi);
        v16bf bm = load_b(xrow + c0, hi);
        acc = __builtin_amdgcn_wmma_f32_16x16x32_bf16(false, a, false, bm,
                                                      (short)0, acc, false, false);
    }
    const int n = n0 + lm;
    if (TRANSPOSED) {
        // [B,N,O]; r and r+1 hit adjacent o -> packed b32 stores
#pragma unroll
        for (int r = 0; r < 8; r += 2) {
            int o = o0 + r + 8 * hi;
            v2bf p = { (bf16_t)(acc[r] + bias[o] * bscale),
                       (bf16_t)(acc[r + 1] + bias[o + 1] * bscale) };
            *reinterpret_cast<v2bf*>(out + ((size_t)b * NPIX + n) * O + o) = p;
        }
    } else {
        // [B,O,N]
#pragma unroll
        for (int r = 0; r < 8; ++r) {
            int o = o0 + r + 8 * hi;
            out[((size_t)b * O + o) * NPIX + n] = (bf16_t)(acc[r] + bias[o] * bscale);
        }
    }
}

// ---- online-softmax step: scores (transposed layout) -> P operand ----------
// Scores arrive pre-scaled (1/sqrt(dk) folded into Wq/bq).
__device__ __forceinline__ void softmax_step(const v8f& s0, const v8f& s1, int hi,
                                             float& m_run, float& l_run,
                                             float& factor, v16bf& p_op) {
    float cmax = -1e30f;
#pragma unroll
    for (int r = 0; r < 8; ++r) cmax = fmaxf(cmax, fmaxf(s0[r], s1[r]));
    cmax = fmaxf(cmax, __shfl_xor(cmax, 16, 32));
    const float m_new = fmaxf(m_run, cmax);
    factor = __expf(m_run - m_new);
    m_run = m_new;
    float psum = 0.f;
#pragma unroll
    for (int r = 0; r < 8; ++r) {
        float p0 = __expf(s0[r] - m_new);   // key-tile0, row r+8*hi
        float p1 = __expf(s1[r] - m_new);   // key-tile1, row r+8*hi
        psum += p0 + p1;
        float send = hi ? p0 : p1;          // exchange halves for B-operand layout
        float recv = __shfl_xor(send, 16, 32);
        p_op[r]     = (bf16_t)(hi ? recv : p0);   // K = r + 16*hi
        p_op[r + 8] = (bf16_t)(hi ? p1 : recv);   // K = r + 8 + 16*hi
    }
    l_run = l_run * factor + psum;
}

// one fully-unrolled PV step (compile-time ct -> acc stays in registers)
#define PV_STEP(ct)                                                                        \
    {                                                                                      \
        const v16bf a_v = load_a(Vb + (size_t)((ct) * 16 + lm) * NPIX + mbase, hi);        \
        acc0[ct] = __builtin_amdgcn_wmma_f32_16x16x32_bf16(false, a_v, false, p0,          \
                                                           (short)0, acc0[ct], false, false); \
        acc1[ct] = __builtin_amdgcn_wmma_f32_16x16x32_bf16(false, a_v, false, p1,          \
                                                           (short)0, acc1[ct], false, false); \
    }

// ---- fused flash attention + residual --------------------------------------
// One wave per (batch, head, 32-query tile): two 16-query sub-tiles share every
// K and V operand load.  PV phase is fully unrolled (register-indexed accs) but
// fenced into two halves so the scheduler can't hoist all 8 V tiles at once.
__global__ void __launch_bounds__(32, 1)
attn_wmma(const bf16_t* __restrict__ Qt, const bf16_t* __restrict__ Kt,
          const bf16_t* __restrict__ V, const float* __restrict__ x,
          const float* __restrict__ gamma, float* __restrict__ out) {
    const int lane = threadIdx.x & 31;
    const int lm = lane & 15, hi = lane >> 4;
    const int h = blockIdx.y, b = blockIdx.z;
    const int n_glob0 = blockIdx.x * 32 + lm;
    const int n_glob1 = n_glob0 + 16;

    const v16bf b_q0 = load_b(Qt + ((size_t)b * NPIX + n_glob0) * DQK + h * DQH, hi);
    const v16bf b_q1 = load_b(Qt + ((size_t)b * NPIX + n_glob1) * DQK + h * DQH, hi);

    const bf16_t* Ktb = Kt + (size_t)b * NPIX * DQK + h * DQH;
    const bf16_t* Vb  = V + ((size_t)b * CDIM + h * DVH) * NPIX;

    v8f acc0[8], acc1[8];
#pragma unroll
    for (int i = 0; i < 8; ++i) { acc0[i] = (v8f){}; acc1[i] = (v8f){}; }
    float m0 = -1e30f, l0 = 0.f, m1 = -1e30f, l1 = 0.f;

    for (int mbase = 0; mbase < NPIX; mbase += 32) {
        if (mbase + 32 < NPIX)   // prefetch next chunk of K rows
            __builtin_prefetch(Ktb + (size_t)(mbase + 32 + lane) * DQK, 0, 0);

        const v16bf a_k0 = load_a(Ktb + (size_t)(mbase + lm) * DQK, hi);
        const v16bf a_k1 = load_a(Ktb + (size_t)(mbase + 16 + lm) * DQK, hi);
        v8f s00 = {}, s01 = {}, s10 = {}, s11 = {};
        s00 = __builtin_amdgcn_wmma_f32_16x16x32_bf16(false, a_k0, false, b_q0, (short)0, s00, false, false);
        s01 = __builtin_amdgcn_wmma_f32_16x16x32_bf16(false, a_k1, false, b_q0, (short)0, s01, false, false);
        s10 = __builtin_amdgcn_wmma_f32_16x16x32_bf16(false, a_k0, false, b_q1, (short)0, s10, false, false);
        s11 = __builtin_amdgcn_wmma_f32_16x16x32_bf16(false, a_k1, false, b_q1, (short)0, s11, false, false);

        float f0, f1;
        v16bf p0, p1;
        softmax_step(s00, s01, hi, m0, l0, f0, p0);
        softmax_step(s10, s11, hi, m1, l1, f1, p1);

#pragma unroll
        for (int t = 0; t < 8; ++t)
#pragma unroll
            for (int r = 0; r < 8; ++r) { acc0[t][r] *= f0; acc1[t][r] *= f1; }

        SCHED_FENCE();            // don't hoist V loads above the rescale
        PV_STEP(0) PV_STEP(1) PV_STEP(2) PV_STEP(3)
        SCHED_FENCE();            // cap V tiles in flight at 4
        PV_STEP(4) PV_STEP(5) PV_STEP(6) PV_STEP(7)
    }

    const float inv_l0 = 1.0f / (l0 + __shfl_xor(l0, 16, 32));
    const float inv_l1 = 1.0f / (l1 + __shfl_xor(l1, 16, 32));
    const float g = gamma[0];
#pragma unroll
    for (int ct = 0; ct < 8; ++ct)
#pragma unroll
        for (int r = 0; r < 8; ++r) {
            const int cch = h * DVH + ct * 16 + r + 8 * hi;
            size_t i0 = ((size_t)b * CDIM + cch) * NPIX + n_glob0;
            size_t i1 = ((size_t)b * CDIM + cch) * NPIX + n_glob1;
            out[i0] = g * (acc0[ct][r] * inv_l0) + x[i0];
            out[i1] = g * (acc1[ct][r] * inv_l1) + x[i1];
        }
}

// ---- launch ----------------------------------------------------------------
extern "C" void kernel_launch(void* const* d_in, const int* in_sizes, int n_in,
                              void* d_out, int out_size, void* d_ws, size_t ws_size,
                              hipStream_t stream) {
    (void)in_sizes; (void)n_in; (void)out_size; (void)ws_size;
    const float* x  = (const float*)d_in[0];
    const float* Wq = (const float*)d_in[1];
    const float* bq = (const float*)d_in[2];
    const float* Wk = (const float*)d_in[3];
    const float* bk = (const float*)d_in[4];
    const float* Wv = (const float*)d_in[5];
    const float* bv = (const float*)d_in[6];
    const float* gm = (const float*)d_in[7];
    float* out = (float*)d_out;

    char* ws = (char*)d_ws;
    size_t off = 0;
    auto alloc = [&](size_t bytes) -> char* {
        char* p = ws + off;
        off += (bytes + 255) & ~(size_t)255;
        return p;
    };
    bf16_t* xT  = (bf16_t*)alloc((size_t)BATCH * NPIX * CDIM * 2);  // [B,N,C]
    bf16_t* Wqb = (bf16_t*)alloc((size_t)DQK * CDIM * 2);
    bf16_t* Wkb = (bf16_t*)alloc((size_t)DQK * CDIM * 2);
    bf16_t* Wvb = (bf16_t*)alloc((size_t)CDIM * CDIM * 2);
    bf16_t* Qt  = (bf16_t*)alloc((size_t)BATCH * NPIX * DQK * 2);   // [B,N,128]
    bf16_t* Ktb = (bf16_t*)alloc((size_t)BATCH * NPIX * DQK * 2);   // [B,N,128]
    bf16_t* Vb  = (bf16_t*)alloc((size_t)BATCH * CDIM * NPIX * 2);  // [B,C,N]

    size_t totx = (size_t)BATCH * (CDIM / 2) * NPIX;
    cvt_x_transpose<<<(unsigned)((totx + 255) / 256), 256, 0, stream>>>(x, xT);
    // fold 1/sqrt(dk) into Wq (bq folded via bscale in proj)
    cvt_flat<<<(DQK * CDIM + 255) / 256, 256, 0, stream>>>(Wq, Wqb, DQK * CDIM, INV_SCALE);
    cvt_flat<<<(DQK * CDIM + 255) / 256, 256, 0, stream>>>(Wk, Wkb, DQK * CDIM, 1.0f);
    cvt_flat<<<(CDIM * CDIM + 255) / 256, 256, 0, stream>>>(Wv, Wvb, CDIM * CDIM, 1.0f);

    proj_wmma<true ><<<dim3(NPIX / 16, DQK / 16, BATCH), 32, 0, stream>>>(Wqb, bq, xT, Qt, DQK, INV_SCALE);
    proj_wmma<true ><<<dim3(NPIX / 16, DQK / 16, BATCH), 32, 0, stream>>>(Wkb, bk, xT, Ktb, DQK, 1.0f);
    proj_wmma<false><<<dim3(NPIX / 16, CDIM / 16, BATCH), 32, 0, stream>>>(Wvb, bv, xT, Vb, CDIM, 1.0f);

    attn_wmma<<<dim3(NPIX / 32, NHEADS, BATCH), 32, 0, stream>>>(Qt, Ktb, Vb, x, gm, out);
}
